// DeepseekV4HyperConnection_14259291423071
// MI455X (gfx1250) — compile-verified, matrix operand order checked by reading, not run on hardware
//
#include <hip/hip_runtime.h>
#include <hip/hip_bf16.h>
#include <math.h>

#define HCN   4
#define HDIM  4096
#define SLEN  4096
#define MIXN  24
#define EPS   1e-6f
#define NEPS  1e-6f

typedef __attribute__((ext_vector_type(16))) _Float16     v16h;
typedef __attribute__((ext_vector_type(8)))  float        v8f;
typedef __attribute__((ext_vector_type(4)))  unsigned int u32x4;
typedef __attribute__((ext_vector_type(2)))  unsigned int u32x2;

__device__ __forceinline__ float sigmoidf(float x) { return 1.0f / (1.0f + expf(-x)); }

// issue one per-lane 16B async copy global->LDS (ASYNCcnt tracked)
__device__ __forceinline__ void async_cp16(unsigned lds_off, const void* gsrc) {
  asm volatile("global_load_async_to_lds_b128 %0, %1, off"
               :: "v"(lds_off), "v"((unsigned long long)(size_t)gsrc)
               : "memory");
}
__device__ __forceinline__ void wait_async0() {
  asm volatile("s_wait_asynccnt 0x0" ::: "memory");
}

// f16 scratch views inside d_out: stream-2 slot = w_inner row (f16),
// stream-3 slot = normed row (f16). Each slot is 16KB, we use 8KB.
__device__ __forceinline__ _Float16* wrow_h(float* dout, int e) {
  return reinterpret_cast<_Float16*>(dout + ((size_t)e * HCN + 2) * HDIM);
}
__device__ __forceinline__ _Float16* nrow_h(float* dout, int s) {
  return reinterpret_cast<_Float16*>(dout + ((size_t)s * HCN + 3) * HDIM);
}

// ---------------------------------------------------------------------------
// Kernel 0: convert w_inner (f32) -> f16 rows in d_out stream-2 slots
// ---------------------------------------------------------------------------
__global__ __launch_bounds__(256) void hc_conv_w(
    const float* __restrict__ w, float* __restrict__ dout)
{
  const int idx = blockIdx.x * 256 + threadIdx.x;   // 4096*512 threads
  const int e = idx >> 9;
  const int c = (idx & 511) * 8;
  const float4 v0 = *reinterpret_cast<const float4*>(w + (size_t)e * HDIM + c);
  const float4 v1 = *reinterpret_cast<const float4*>(w + (size_t)e * HDIM + c + 4);
  union { _Float16 h[8]; u32x4 u; } pk;
  pk.h[0] = (_Float16)v0.x; pk.h[1] = (_Float16)v0.y;
  pk.h[2] = (_Float16)v0.z; pk.h[3] = (_Float16)v0.w;
  pk.h[4] = (_Float16)v1.x; pk.h[5] = (_Float16)v1.y;
  pk.h[6] = (_Float16)v1.z; pk.h[7] = (_Float16)v1.w;
  *reinterpret_cast<u32x4*>(wrow_h(dout, e) + c) = pk.u;
}

// ---------------------------------------------------------------------------
// Kernel 1: per-token RMS stat + 24 mixes + gates + Sinkhorn + reduced +
// RMSNorm. One block (256 threads, 8 wave32) per token.
// normed (f16) -> d_out stream-3 slot.
// ---------------------------------------------------------------------------
__global__ __launch_bounds__(256) void hc_gates_norm(
    const float* __restrict__ hs, const float* __restrict__ hc_fn,
    const float* __restrict__ hc_base, const float* __restrict__ hc_scale,
    const float* __restrict__ norm_w, float* __restrict__ dout,
    float* __restrict__ postg, float* __restrict__ combg)
{
  const int s   = blockIdx.x;
  const int tid = threadIdx.x;
  const float* row = hs + (size_t)s * (HCN * HDIM);

  __shared__ float red[25 * 8];
  __shared__ float pre_s[HCN];
  __shared__ float nr_s;
  __shared__ float buf[HDIM];   // 16 KB: holds "reduced"

  // ---- phase 1: ssq + 24 dot products over 16384 features ----
  float ssq = 0.f;
  float m[MIXN];
  #pragma unroll
  for (int j = 0; j < MIXN; ++j) m[j] = 0.f;

  for (int it = 0; it < 16; ++it) {
    const int i4 = it * 256 + tid;
    const float4 v = reinterpret_cast<const float4*>(row)[i4];
    ssq += v.x * v.x + v.y * v.y + v.z * v.z + v.w * v.w;
    #pragma unroll
    for (int j = 0; j < MIXN; ++j) {
      const float4 w = reinterpret_cast<const float4*>(hc_fn + (size_t)j * (HCN * HDIM))[i4];
      m[j] += v.x * w.x + v.y * w.y + v.z * w.z + v.w * w.w;
    }
  }

  #pragma unroll
  for (int off = 16; off > 0; off >>= 1) {
    ssq += __shfl_xor(ssq, off, 32);
    #pragma unroll
    for (int j = 0; j < MIXN; ++j) m[j] += __shfl_xor(m[j], off, 32);
  }
  const int wv = tid >> 5, ln = tid & 31;
  if (ln == 0) {
    red[0 * 8 + wv] = ssq;
    #pragma unroll
    for (int j = 0; j < MIXN; ++j) red[(1 + j) * 8 + wv] = m[j];
  }
  __syncthreads();

  // ---- gates + Sinkhorn on thread 0 ----
  if (tid == 0) {
    float tot[25];
    #pragma unroll
    for (int q = 0; q < 25; ++q) {
      float t = 0.f;
      #pragma unroll
      for (int w = 0; w < 8; ++w) t += red[q * 8 + w];
      tot[q] = t;
    }
    const float rs = rsqrtf(tot[0] * (1.0f / (float)(HCN * HDIM)) + NEPS);
    const float ps = hc_scale[0], qs = hc_scale[1], cs = hc_scale[2];
    float post[4], c[4][4];
    #pragma unroll
    for (int h = 0; h < 4; ++h) {
      pre_s[h] = sigmoidf(tot[1 + h]     * rs * ps + hc_base[h])     + EPS;
      post[h]  = sigmoidf(tot[1 + 4 + h] * rs * qs + hc_base[4 + h]) + EPS;
    }
    #pragma unroll
    for (int h = 0; h < 4; ++h)
      #pragma unroll
      for (int k = 0; k < 4; ++k)
        c[h][k] = sigmoidf(tot[1 + 8 + h * 4 + k] * rs * cs + hc_base[8 + h * 4 + k]) + EPS;

    #pragma unroll 1
    for (int it = 0; it < 20; ++it) {
      #pragma unroll
      for (int h = 0; h < 4; ++h) {
        const float inv = 1.0f / (c[h][0] + c[h][1] + c[h][2] + c[h][3] + EPS);
        #pragma unroll
        for (int k = 0; k < 4; ++k) c[h][k] *= inv;
      }
      #pragma unroll
      for (int k = 0; k < 4; ++k) {
        const float inv = 1.0f / (c[0][k] + c[1][k] + c[2][k] + c[3][k] + EPS);
        #pragma unroll
        for (int h = 0; h < 4; ++h) c[h][k] *= inv;
      }
    }
    #pragma unroll
    for (int h = 0; h < 4; ++h) postg[s * 4 + h] = post[h];
    #pragma unroll
    for (int h = 0; h < 4; ++h)
      #pragma unroll
      for (int k = 0; k < 4; ++k) combg[s * 16 + h * 4 + k] = c[h][k];
  }
  __syncthreads();

  // ---- phase 2: reduced = sum_h pre[h]*hidden; RMSNorm -> f16 normed ----
  const float p0 = pre_s[0], p1 = pre_s[1], p2 = pre_s[2], p3 = pre_s[3];
  float ssq2 = 0.f;
  for (int it = 0; it < 4; ++it) {
    const int d4 = it * 256 + tid;
    const float4 a  = reinterpret_cast<const float4*>(row)[d4];
    const float4 b  = reinterpret_cast<const float4*>(row + HDIM)[d4];
    const float4 cc = reinterpret_cast<const float4*>(row + 2 * HDIM)[d4];
    const float4 dd = reinterpret_cast<const float4*>(row + 3 * HDIM)[d4];
    float4 r;
    r.x = p0 * a.x + p1 * b.x + p2 * cc.x + p3 * dd.x;
    r.y = p0 * a.y + p1 * b.y + p2 * cc.y + p3 * dd.y;
    r.z = p0 * a.z + p1 * b.z + p2 * cc.z + p3 * dd.z;
    r.w = p0 * a.w + p1 * b.w + p2 * cc.w + p3 * dd.w;
    ssq2 += r.x * r.x + r.y * r.y + r.z * r.z + r.w * r.w;
    reinterpret_cast<float4*>(buf)[d4] = r;
  }
  #pragma unroll
  for (int off = 16; off > 0; off >>= 1) ssq2 += __shfl_xor(ssq2, off, 32);
  if (ln == 0) red[wv] = ssq2;
  __syncthreads();
  if (tid == 0) {
    float t = 0.f;
    #pragma unroll
    for (int w = 0; w < 8; ++w) t += red[w];
    nr_s = rsqrtf(t * (1.0f / (float)HDIM) + NEPS);
  }
  __syncthreads();

  const float nr = nr_s;
  _Float16* nh = nrow_h(dout, s);
  for (int it = 0; it < 4; ++it) {
    const int d4 = it * 256 + tid;
    const float4 r = reinterpret_cast<const float4*>(buf)[d4];
    const float4 w = reinterpret_cast<const float4*>(norm_w)[d4];
    union { _Float16 h[4]; u32x2 u; } pk;
    pk.h[0] = (_Float16)(r.x * nr * w.x);
    pk.h[1] = (_Float16)(r.y * nr * w.y);
    pk.h[2] = (_Float16)(r.z * nr * w.z);
    pk.h[3] = (_Float16)(r.w * nr * w.w);
    *reinterpret_cast<u32x2*>(&nh[d4 * 4]) = pk.u;
  }
}

// ---------------------------------------------------------------------------
// Kernel 2: proj[s][e] = sum_d normed[s][d] * w_inner[e][d]
// WMMA f32<-f16 GEMM. 128x128 block tile, K-step 32, 8 wave32/block.
// Tiles staged with global_load_async_to_lds_b128, double-buffered,
// one barrier per K-step. proj -> d_out stream-0 slots.
// ---------------------------------------------------------------------------
#define TM 128
#define TN 128
#define TK 32
#define LDT 40   // padded half-stride: 80B rows, 16B-aligned fragment loads

union Frag { v16h h; u32x4 q[2]; };

__global__ __launch_bounds__(256) void hc_gemm(float* __restrict__ dout)
{
  __shared__ __align__(16) _Float16 lA[2][TM * LDT];
  __shared__ __align__(16) _Float16 lB[2][TN * LDT];

  const int tid = threadIdx.x;
  const int s0  = blockIdx.y * TM;   // token rows
  const int n0  = blockIdx.x * TN;   // output feature cols
  const int wv  = tid >> 5;
  const int ln  = tid & 31;
  const int wm  = (wv & 3) * 32;     // wave M offset
  const int wn  = (wv >> 2) * 64;    // wave N offset
  const int hl  = ln >> 4;
  const int lr  = ln & 15;

  const unsigned baseA = (unsigned)(size_t)&lA[0][0];
  const unsigned baseB = (unsigned)(size_t)&lB[0][0];

  // per-lane copy slot: 8 waves x (2 x 8 rows) cover 128 rows; 4 lanes/row x 16B
  const int cr  = (ln >> 2);          // row-within-8
  const int ckg = (ln & 3) * 8;       // half offset 0,8,16,24

  const v8f z8 = {0.f, 0.f, 0.f, 0.f, 0.f, 0.f, 0.f, 0.f};
  v8f acc[2][4];
  #pragma unroll
  for (int i = 0; i < 2; ++i)
    #pragma unroll
    for (int j = 0; j < 4; ++j) acc[i][j] = z8;

  auto issueTile = [&](int kb, int bsel) {
    #pragma unroll
    for (int i = 0; i < 2; ++i) {
      const int r = wv * 16 + i * 8 + cr;
      const unsigned loff = (unsigned)((bsel * (TM * LDT) + r * LDT + ckg) * 2);
      async_cp16(baseA + loff, nrow_h(dout, s0 + r) + kb + ckg);
      async_cp16(baseB + loff, wrow_h(dout, n0 + r) + kb + ckg);
    }
  };

  issueTile(0, 0);
  wait_async0();
  __syncthreads();

  int cur = 0;
  for (int kb = 0; kb < HDIM; kb += TK) {
    if (kb + TK < HDIM) issueTile(kb + TK, cur ^ 1);

    // fragment loads per ISA §7.12.2 16-bit A layout
    Frag a[2], b[4];
    #pragma unroll
    for (int i = 0; i < 2; ++i) {
      const _Float16* p = &lA[cur][(wm + i * 16 + lr) * LDT];
      a[i].q[0] = *reinterpret_cast<const u32x4*>(p + hl * 8);
      a[i].q[1] = *reinterpret_cast<const u32x4*>(p + 16 + hl * 8);
    }
    #pragma unroll
    for (int j = 0; j < 4; ++j) {
      const _Float16* p = &lB[cur][(wn + j * 16 + lr) * LDT];
      b[j].q[0] = *reinterpret_cast<const u32x4*>(p + hl * 8);
      b[j].q[1] = *reinterpret_cast<const u32x4*>(p + 16 + hl * 8);
    }

    #pragma unroll
    for (int i = 0; i < 2; ++i)
      #pragma unroll
      for (int j = 0; j < 4; ++j)
        acc[i][j] = __builtin_amdgcn_wmma_f32_16x16x32_f16(
            false, a[i].h, false, b[j].h, (short)0, acc[i][j], false, false);

    wait_async0();     // next tile landed (own copies)
    __syncthreads();   // everyone done reading cur + all copies visible
    cur ^= 1;
  }

  // epilogue: C layout -> proj at d_out stream-0 slots
  #pragma unroll
  for (int i = 0; i < 2; ++i) {
    #pragma unroll
    for (int j = 0; j < 4; ++j) {
      const int sRow = s0 + wm + i * 16 + hl * 8;
      const int e    = n0 + wn + j * 16 + lr;
      #pragma unroll
      for (int g = 0; g < 8; ++g)
        dout[((size_t)(sRow + g) * HCN + 0) * HDIM + e] = acc[i][j][g];
    }
  }
}

// ---------------------------------------------------------------------------
// Kernel 3: expanded[s][h][d] = post[s][h]*proj[s][d] + sum_k comb[s][h][k]*hid
// One thread per (s,d); reads proj from d_out[s][0][d], writes all 4 slots
// (clobbering the f16 scratch in slots 2/3, which is no longer needed).
// ---------------------------------------------------------------------------
__global__ __launch_bounds__(256) void hc_combine(
    const float* __restrict__ hs, const float* __restrict__ postg,
    const float* __restrict__ combg, float* __restrict__ dout)
{
  const int idx = blockIdx.x * 256 + threadIdx.x;
  const int s = idx >> 12;
  const int d = idx & (HDIM - 1);

  const float4 po = *reinterpret_cast<const float4*>(postg + s * 4);
  const float4 c0 = *reinterpret_cast<const float4*>(combg + s * 16 + 0);
  const float4 c1 = *reinterpret_cast<const float4*>(combg + s * 16 + 4);
  const float4 c2 = *reinterpret_cast<const float4*>(combg + s * 16 + 8);
  const float4 c3 = *reinterpret_cast<const float4*>(combg + s * 16 + 12);

  const size_t base = (size_t)s * HCN * HDIM + d;
  const float h0 = hs[base];
  const float h1 = hs[base + HDIM];
  const float h2 = hs[base + 2 * HDIM];
  const float h3 = hs[base + 3 * HDIM];
  const float p  = dout[base];   // proj (stream-0 scratch), same-thread RMW

  const float o0 = po.x * p + c0.x * h0 + c0.y * h1 + c0.z * h2 + c0.w * h3;
  const float o1 = po.y * p + c1.x * h0 + c1.y * h1 + c1.z * h2 + c1.w * h3;
  const float o2 = po.z * p + c2.x * h0 + c2.y * h1 + c2.z * h2 + c2.w * h3;
  const float o3 = po.w * p + c3.x * h0 + c3.y * h1 + c3.z * h2 + c3.w * h3;

  dout[base]            = o0;
  dout[base + HDIM]     = o1;
  dout[base + 2 * HDIM] = o2;
  dout[base + 3 * HDIM] = o3;
}

// ---------------------------------------------------------------------------
extern "C" void kernel_launch(void* const* d_in, const int* in_sizes, int n_in,
                              void* d_out, int out_size, void* d_ws, size_t ws_size,
                              hipStream_t stream) {
  (void)in_sizes; (void)n_in; (void)out_size; (void)ws_size;
  const float* hs       = (const float*)d_in[0];
  const float* hc_fn    = (const float*)d_in[1];
  const float* hc_base  = (const float*)d_in[2];
  const float* hc_scale = (const float*)d_in[3];
  const float* norm_w   = (const float*)d_in[4];
  const float* w_inner  = (const float*)d_in[5];
  float* dout  = (float*)d_out;
  float* postg = (float*)d_ws;                  // 4096*4 floats
  float* combg = postg + (size_t)SLEN * 4;      // 4096*16 floats (ws need: 320 KB)

  hc_conv_w<<<(SLEN * 512) / 256, 256, 0, stream>>>(w_inner, dout);
  hc_gates_norm<<<SLEN, 256, 0, stream>>>(hs, hc_fn, hc_base, hc_scale,
                                          norm_w, dout, postg, combg);
  hc_gemm<<<dim3(HDIM / TN, SLEN / TM), 256, 0, stream>>>(dout);
  hc_combine<<<(SLEN * HDIM) / 256, 256, 0, stream>>>(hs, postg, combg, dout);
}
